// ElmanNLM_4690104287243
// MI455X (gfx1250) — matrix operand order repaired and checked by reading, not compiled, not run on hardware
//
#include <hip/hip_runtime.h>
#include <hip/hip_bf16.h>
#include <math.h>

#define VOCAB 32000
#define HID   256
#define BATCH 8
#define SEQ   256
#define MROWS (BATCH * SEQ)   // 2048
#define KSPLIT 8
#define KSEG  (VOCAB / KSPLIT) // 4000

typedef __attribute__((ext_vector_type(16))) __bf16 v16bf;
typedef __attribute__((ext_vector_type(8)))  float  v8f;

// ---- fragment loaders (CDNA5 wave32 WMMA 16x16x32 bf16 layouts) -------------
// A (16x32, MxK): lane&15 = row M; lanes 0-15 hold K 0..15 in elems
// 0..15; lanes 16-31 hold K 16..31.
static __device__ __forceinline__ v16bf a_frag_from_f32(const float* __restrict__ rowp,
                                                        int kc, int lane) {
  int k0 = kc + ((lane & 16) ? 16 : 0);
  int k1 = kc + ((lane & 16) ? 24 : 8);
  v16bf a;
#pragma unroll
  for (int i = 0; i < 8; ++i) a[i] = (__bf16)rowp[k0 + i];
#pragma unroll
  for (int i = 0; i < 8; ++i) a[i + 8] = (__bf16)rowp[k1 + i];
  return a;
}

static __device__ __forceinline__ v16bf a_frag_from_bf16(const __bf16* __restrict__ rowp,
                                                         int kc, int lane) {
  int k0 = kc + ((lane & 16) ? 16 : 0);
  int k1 = kc + ((lane & 16) ? 24 : 8);
  v16bf a;
#pragma unroll
  for (int i = 0; i < 8; ++i) a[i] = rowp[k0 + i];
#pragma unroll
  for (int i = 0; i < 8; ++i) a[i + 8] = rowp[k1 + i];
  return a;
}

// B (32x16, KxN): lane&15 = column N (baked into colp); lanes 0-15 hold
// K kc..kc+15 (elem e -> K=kc+e), lanes 16-31 hold K kc+16..kc+31.
// bf16 weights: the whole fragment is one contiguous 32-byte load.
static __device__ __forceinline__ v16bf b_frag_from_bf16(const __bf16* __restrict__ colp,
                                                         int kc, int lane) {
  int kk = kc + ((lane & 16) ? 16 : 0);
  return *(const v16bf*)(colp + kk);   // 32B aligned: kk % 16 == 0
}

static __device__ __forceinline__ v16bf b_frag_from_f32(const float* __restrict__ colp,
                                                        int kc, int lane) {
  int kk = kc + ((lane & 16) ? 16 : 0);
  v16bf b;
#pragma unroll
  for (int i = 0; i < 16; ++i) b[i] = (__bf16)colp[kk + i];
  return b;
}

// ---- stage -1: one-time f32 -> bf16 weight down-convert into workspace -----
__global__ __launch_bounds__(256) void k_f32_to_bf16(const float* __restrict__ src,
                                                     __bf16* __restrict__ dst, int n) {
  int i = (blockIdx.x * 256 + threadIdx.x) * 8;
  if (i + 8 <= n) {
#pragma unroll
    for (int j = 0; j < 8; ++j) dst[i + j] = (__bf16)src[i + j];
  }
}

// ---- stage 0: x_pre = b_ih + b_hh (split-K GEMM accumulates on top) --------
__global__ __launch_bounds__(256) void k_bias_init(const float* __restrict__ b_ih,
                                                   const float* __restrict__ b_hh,
                                                   float* __restrict__ xpre) {
  int i = blockIdx.x * 256 + threadIdx.x;
  if (i < MROWS * HID) xpre[i] = b_ih[i & (HID - 1)] + b_hh[i & (HID - 1)];
}

// ---- stage 1: x_pre += input_seq @ W_ih^T   (M=2048, N=256, K=32000) -------
// Block = 4 waves; each wave owns 4 N-tiles (covers all H=256 per block) so the
// A fragment is loaded+converted once per K-chunk and reused 4x. B fragments
// are raw bf16 b128 loads (no VALU after the WMMA -> no hazard NOPs). Split-K
// over grid.y with f32 atomic accumulation (W_ih bf16 is fully L2-resident).
__global__ __launch_bounds__(128) void k_gemm1(const float* __restrict__ X,
                                               const __bf16* __restrict__ Wih,
                                               float* __restrict__ xpre) {
  int lane = threadIdx.x & 31;
  int wave = threadIdx.x >> 5;  // 0..3
  int m0 = blockIdx.x * 16;
  int mrow = m0 + (lane & 15);
  int kbeg = blockIdx.y * KSEG;
  const float* rowp = X + (size_t)mrow * VOCAB;

  int nn[4];
  const __bf16* colp[4];
  v8f acc[4];
#pragma unroll
  for (int j = 0; j < 4; ++j) {
    nn[j] = j * 64 + wave * 16 + (lane & 15);
    colp[j] = Wih + (size_t)nn[j] * VOCAB;
    acc[j] = (v8f){};
  }

  for (int kc = kbeg; kc < kbeg + KSEG; kc += 32) {
    __builtin_prefetch(rowp + kc + 256, 0, 1);       // global_prefetch_b8
    __builtin_prefetch(colp[0] + kc + 512, 0, 1);
    v16bf a = a_frag_from_f32(rowp, kc, lane);
#pragma unroll
    for (int j = 0; j < 4; ++j) {
      v16bf b = b_frag_from_bf16(colp[j], kc, lane);
      acc[j] = __builtin_amdgcn_wmma_f32_16x16x32_bf16(false, a, false, b,
                                                       (short)0, acc[j], false, false);
    }
  }

  int hi = lane >> 4;  // C layout: VGPR r -> row r + 8*hi, col = lane&15
#pragma unroll
  for (int j = 0; j < 4; ++j)
#pragma unroll
    for (int r = 0; r < 8; ++r) {
      int m = m0 + r + 8 * hi;
      atomicAdd(&xpre[(size_t)m * HID + nn[j]], acc[j][r]);
    }
}

// ---- stage 2: sequential Elman scan, one workgroup, 16 waves ---------------
// h held in LDS as bf16 (batch padded 8->16, pad rows stay zero); W_hh held in
// registers as 8 pre-converted bf16 B-fragments per wave (wave owns 16 of 256 H).
__global__ __launch_bounds__(512) void k_rnn(const float* __restrict__ h0,
                                             const float* __restrict__ Whh,
                                             const float* __restrict__ xpre,
                                             __bf16* __restrict__ hs,
                                             float* __restrict__ hlast) {
  __shared__ __bf16 hsh[16 * HID];
  int tid = threadIdx.x;
  int lane = tid & 31;
  int wave = tid >> 5;  // 0..15

  for (int i = tid; i < 16 * HID; i += 512) {
    int b = i >> 8;  // / HID
    hsh[i] = (__bf16)((b < BATCH) ? h0[i] : 0.0f);
  }

  int n = wave * 16 + (lane & 15);
  int hi = lane >> 4;
  v16bf bw[8];
  const float* colp = Whh + (size_t)n * HID;
#pragma unroll
  for (int c = 0; c < 8; ++c) bw[c] = b_frag_from_f32(colp, c * 32, lane);
  __syncthreads();

  const __bf16* arow = hsh + (size_t)(lane & 15) * HID;
  for (int t = 0; t < SEQ; ++t) {
    v8f acc = {};
    if (hi == 0) {  // rows 0..7 are real batch rows; rows 8..15 padding stay 0
#pragma unroll
      for (int r = 0; r < 8; ++r)
        acc[r] = xpre[((size_t)r * SEQ + t) * HID + n];
    }
#pragma unroll
    for (int c = 0; c < 8; ++c) {
      v16bf a = a_frag_from_bf16(arow, c * 32, lane);
      acc = __builtin_amdgcn_wmma_f32_16x16x32_bf16(false, a, false, bw[c],
                                                    (short)0, acc, false, false);
    }
    __syncthreads();  // all waves done reading old h
    if (hi == 0) {
#pragma unroll
      for (int r = 0; r < 8; ++r) {
        float v = tanhf(acc[r]);
        hsh[r * HID + n] = (__bf16)v;
        hs[((size_t)r * SEQ + t) * HID + n] = (__bf16)v;
        if (t == SEQ - 1) hlast[r * HID + n] = v;  // fp32 h_last
      }
    }
    __syncthreads();  // new h visible
  }
}

// ---- stage 3: logits = hs @ W_fc^T + b_fc  (M=2048, N=32000, K=256) --------
// A and B both bf16 in ws: inner loop is pure b128 loads + WMMA.
__global__ __launch_bounds__(256) void k_gemm2(const __bf16* __restrict__ hs,
                                               const __bf16* __restrict__ Wfc,
                                               const float* __restrict__ bfc,
                                               float* __restrict__ out) {
  int lane = threadIdx.x & 31;
  int wave = threadIdx.x >> 5;  // 0..7
  int m0 = blockIdx.x * 16;
  int n = blockIdx.y * 128 + wave * 16 + (lane & 15);
  const __bf16* arow = hs + (size_t)(m0 + (lane & 15)) * HID;
  const __bf16* colp = Wfc + (size_t)n * HID;

  v8f acc = {};
#pragma unroll
  for (int c = 0; c < 8; ++c) {
    v16bf a = a_frag_from_bf16(arow, c * 32, lane);
    v16bf b = b_frag_from_bf16(colp, c * 32, lane);
    acc = __builtin_amdgcn_wmma_f32_16x16x32_bf16(false, a, false, b,
                                                  (short)0, acc, false, false);
  }
  float bias = bfc[n];
  int hi = lane >> 4;
#pragma unroll
  for (int r = 0; r < 8; ++r)
    out[(size_t)(m0 + r + 8 * hi) * VOCAB + n] = acc[r] + bias;
}

// ---- stage 4: in-place log-softmax over V, one workgroup per row -----------
__global__ __launch_bounds__(256) void k_logsoftmax(float* __restrict__ out) {
  __shared__ float rm[256], rs[256];
  float* p = out + (size_t)blockIdx.x * VOCAB;
  int tid = threadIdx.x;

  float m = -INFINITY, s = 0.0f;  // online (max, sumexp)
  for (int i = tid; i < VOCAB; i += 256) {
    float x = p[i];
    if (x > m) { s = s * expf(m - x) + 1.0f; m = x; }
    else       { s += expf(x - m); }
  }
  rm[tid] = m; rs[tid] = s;
  __syncthreads();
  for (int off = 128; off > 0; off >>= 1) {
    if (tid < off) {
      float m2 = rm[tid + off], s2 = rs[tid + off];
      float M = fmaxf(rm[tid], m2);
      rs[tid] = rs[tid] * expf(rm[tid] - M) + s2 * expf(m2 - M);
      rm[tid] = M;
    }
    __syncthreads();
  }
  float L = rm[0] + logf(rs[0]);
  for (int i = tid; i < VOCAB; i += 256) p[i] -= L;
}

extern "C" void kernel_launch(void* const* d_in, const int* in_sizes, int n_in,
                              void* d_out, int out_size, void* d_ws, size_t ws_size,
                              hipStream_t stream) {
  (void)in_sizes; (void)n_in; (void)out_size; (void)ws_size;
  const float* input_seq = (const float*)d_in[0];  // [B,T,V]
  const float* h0        = (const float*)d_in[1];  // [1,B,H]
  const float* W_ih      = (const float*)d_in[2];  // [H,V]
  const float* W_hh      = (const float*)d_in[3];  // [H,H]
  const float* b_ih      = (const float*)d_in[4];  // [H]
  const float* b_hh      = (const float*)d_in[5];  // [H]
  const float* W_fc      = (const float*)d_in[6];  // [V,H]
  const float* b_fc      = (const float*)d_in[7];  // [V]

  float* out   = (float*)d_out;                        // [B,T,V] log_probs
  float* hlast = out + (size_t)MROWS * VOCAB;          // [1,B,H]

  // workspace layout (~34.3 MB total)
  char* ws = (char*)d_ws;
  float*  xpre  = (float*)ws;                                   // 2 MB
  ws += (size_t)MROWS * HID * sizeof(float);
  __bf16* hs    = (__bf16*)ws;                                  // 1 MB
  ws += (size_t)MROWS * HID * sizeof(__bf16);
  __bf16* wih16 = (__bf16*)ws;                                  // 16 MB
  ws += (size_t)HID * VOCAB * sizeof(__bf16);
  __bf16* wfc16 = (__bf16*)ws;                                  // 16 MB

  const int WN = HID * VOCAB;  // 8,192,000 elements (multiple of 8)
  k_f32_to_bf16<<<dim3(WN / 8 / 256), dim3(256), 0, stream>>>(W_ih, wih16, WN);
  k_f32_to_bf16<<<dim3(WN / 8 / 256), dim3(256), 0, stream>>>(W_fc, wfc16, WN);

  k_bias_init <<<dim3((MROWS * HID + 255) / 256), dim3(256), 0, stream>>>(b_ih, b_hh, xpre);
  k_gemm1     <<<dim3(MROWS / 16, KSPLIT),        dim3(128), 0, stream>>>(input_seq, wih16, xpre);
  k_rnn       <<<dim3(1),                         dim3(512), 0, stream>>>(h0, W_hh, xpre, hs, hlast);
  k_gemm2     <<<dim3(MROWS / 16, VOCAB / 128),   dim3(256), 0, stream>>>(hs, wfc16, b_fc, out);
  k_logsoftmax<<<dim3(MROWS),                     dim3(256), 0, stream>>>(out);
}
